// MHSA_42863773614629
// MI455X (gfx1250) — matrix operand order repaired and checked by reading, not compiled, not run on hardware
//
#include <hip/hip_runtime.h>
#include <hip/hip_bf16.h>

typedef __attribute__((ext_vector_type(16))) __bf16       v16bf;
typedef __attribute__((ext_vector_type(8)))  float        v8f;
typedef __attribute__((ext_vector_type(4)))  unsigned int v4u;
typedef __attribute__((ext_vector_type(4)))  float        v4f;
typedef __attribute__((ext_vector_type(8)))  int          v8i;
typedef __attribute__((ext_vector_type(4)))  int          v4i;

#define HEADS 4
#define DQK   128
#define DV    128
#define BB    16
#define CC    512
#define NN    1024
#define SCALE 0.08838834764831845f   // 128^-0.5

// padded LDS row pitches (halves) -> conflict-free ds_load_b128 fragment reads
#define KT_PITCH 136   // 128 + 8  (68 dwords/row)
#define VT_PITCH 72    // 64 + 8   (36 dwords/row)
#define PL_PITCH 72    // 64 + 8

static __device__ __forceinline__ unsigned short f2bf(float f) {
  unsigned u = __float_as_uint(f);
  u += 0x7FFFu + ((u >> 16) & 1u);       // round-to-nearest-even
  return (unsigned short)(u >> 16);
}

union Frag { v4u u[2]; v16bf v; };

static __device__ __forceinline__ v8f wmma_bf16(v16bf a, v16bf b, v8f c) {
  return __builtin_amdgcn_wmma_f32_16x16x32_bf16(false, a, false, b, (short)0, c,
                                                 false, false);
}

// A-fragment: operand [row][K], K contiguous. lane l: halves k0+(l>>4)*8+{0..7}, +16..23
static __device__ __forceinline__ v16bf load_a(const unsigned short* rowptr, int k0, int lane) {
  Frag f;
  int base = k0 + ((lane >> 4) << 3);
  f.u[0] = *(const v4u*)(rowptr + base);
  f.u[1] = *(const v4u*)(rowptr + base + 16);
  return f.v;
}

// B-fragment: operand [col][K], K contiguous. lane l: halves k0+(l>>4)*8+{0..7}, +16..23
static __device__ __forceinline__ v16bf load_b(const unsigned short* colptr, int k0, int lane) {
  Frag f;
  int base = k0 + ((lane >> 4) << 3);
  f.u[0] = *(const v4u*)(colptr + base);
  f.u[1] = *(const v4u*)(colptr + base + 16);
  return f.v;
}

// ---------------- Tensor Data Mover: 2-D tile global -> LDS with row padding -----
#if __has_builtin(__builtin_amdgcn_tensor_load_to_lds)
#define HAVE_TDM 1
static __device__ __forceinline__ void tdm_load_2d(
    unsigned lds_addr, const void* gptr,
    unsigned tile_d0, unsigned tile_d1,          // elements / rows
    unsigned long long stride0,                  // row pitch, elements
    unsigned tensor_d0, unsigned tensor_d1,
    unsigned pad_interval_code, unsigned pad_amount_code) {
  unsigned long long ga = (unsigned long long)(uintptr_t)gptr;
  v4u g0;
  g0[0] = 1u;                                               // count=1, user mode
  g0[1] = lds_addr;                                         // lds_addr[31:0]
  g0[2] = (unsigned)ga;                                     // global_addr[31:0]
  g0[3] = (unsigned)((ga >> 32) & 0x1FFFFFFu) | (2u << 30); // ga[56:32] | type=2
  v8i g1;
  g1[0] = (int)((1u << 16) | (1u << 20) |                   // data_size=2B, pad_enable
                (pad_interval_code << 22) | (pad_amount_code << 25));
  g1[1] = (int)((tensor_d0 & 0xFFFFu) << 16);               // tensor_dim0[15:0]
  g1[2] = (int)(((tensor_d0 >> 16) & 0xFFFFu) | ((tensor_d1 & 0xFFFFu) << 16));
  g1[3] = (int)(((tensor_d1 >> 16) & 0xFFFFu) | ((tile_d0 & 0xFFFFu) << 16));
  g1[4] = (int)(tile_d1 & 0xFFFFu);                         // tile_dim1, tile_dim2=0
  g1[5] = (int)(unsigned)(stride0 & 0xFFFFFFFFull);         // tensor_dim0_stride lo
  g1[6] = (int)(unsigned)((stride0 >> 32) & 0xFFFFull);     // stride hi, dim1_stride=0
  g1[7] = 0;
  v4i z4 = {0, 0, 0, 0};
#if defined(__clang_major__) && __clang_major__ >= 23
  v8i z8 = {0, 0, 0, 0, 0, 0, 0, 0};
  __builtin_amdgcn_tensor_load_to_lds(g0, g1, z4, z4, z8, 0);
#else
  __builtin_amdgcn_tensor_load_to_lds(g0, g1, z4, z4, 0);
#endif
}
#else
#define HAVE_TDM 0
#endif

// ---------------- prep: featuremap [B,C,N] f32 -> Xt [B,N,C] bf16 ----------------
__global__ void k_transpose(const float* __restrict__ fm, unsigned short* __restrict__ Xt) {
  __shared__ float tile[32][33];
  int tx = threadIdx.x, ty = threadIdx.y;
  int n0 = blockIdx.x * 32, c0 = blockIdx.y * 32, b = blockIdx.z;
  const float* src = fm + (size_t)b * CC * NN;
#pragma unroll
  for (int i = 0; i < 4; ++i)
    tile[ty + 8 * i][tx] = src[(size_t)(c0 + ty + 8 * i) * NN + n0 + tx];
  __syncthreads();
#pragma unroll
  for (int i = 0; i < 4; ++i) {
    int n = n0 + ty + 8 * i;
    Xt[((size_t)b * NN + n) * CC + c0 + tx] = f2bf(tile[tx][ty + 8 * i]);
  }
}

// ---------------- prep: weights f32 -> bf16 ----------------
__global__ void k_wconv(const float* __restrict__ wqk, const float* __restrict__ wv,
                        unsigned short* __restrict__ oqk, unsigned short* __restrict__ ov) {
  int i = blockIdx.x * 256 + threadIdx.x;
  if (i < 1024 * 512) oqk[i] = f2bf(wqk[i]);
  else                ov[i - 1024 * 512] = f2bf(wv[i - 1024 * 512]);
}

// ---------------- projection: Q (scaled), K' (=K+emb), V^T ----------------
__global__ void k_proj(const unsigned short* __restrict__ Xt,
                       const unsigned short* __restrict__ Wqk,
                       const unsigned short* __restrict__ Wv,
                       const float* __restrict__ emb_h,
                       const float* __restrict__ emb_w,
                       unsigned short* __restrict__ Qs,
                       unsigned short* __restrict__ Kp,
                       unsigned short* __restrict__ Vt) {
  int lane = threadIdx.x & 31;
  int wid  = blockIdx.x * 8 + (threadIdx.x >> 5);
  int b    = wid / (96 * 16);
  int rem  = wid % (96 * 16);
  int ot   = rem >> 4;            // 0..95
  int n0   = (rem & 15) * 64;
  int col  = lane & 15;
  int half8 = (lane >> 4) << 3;

  const unsigned short* wbase;
  int orow0;
  if (ot < 64) { wbase = Wqk; orow0 = ot * 16; }
  else         { wbase = Wv;  orow0 = (ot - 64) * 16; }
  const unsigned short* arow = wbase + (size_t)(orow0 + col) * CC;
  const unsigned short* xb   = Xt + (size_t)b * NN * CC;

  v8f acc[4] = {};
  for (int kt = 0; kt < 16; ++kt) {
    v16bf a = load_a(arow, kt * 32, lane);
#pragma unroll
    for (int t = 0; t < 4; ++t) {
      const unsigned short* brow = xb + (size_t)(n0 + t * 16 + col) * CC;
      v16bf bf = load_b(brow, kt * 32, lane);
      acc[t] = wmma_bf16(a, bf, acc[t]);
    }
  }

#pragma unroll
  for (int t = 0; t < 4; ++t) {
    int n = n0 + t * 16 + col;
    if (ot < 32) {                              // Q: scaled, [bh][n][d]
      int oc = ot * 16, head = oc >> 7, d0 = (oc & 127) + half8;
      v4u pk;
#pragma unroll
      for (int j = 0; j < 4; ++j)
        pk[j] = (unsigned)f2bf(acc[t][2 * j] * SCALE) |
                ((unsigned)f2bf(acc[t][2 * j + 1] * SCALE) << 16);
      *(v4u*)(Qs + ((size_t)(b * HEADS + head) * NN + n) * DQK + d0) = pk;
    } else if (ot < 64) {                       // K': +emb, [bh][n][d]
      int oc = ot * 16 - 512, head = oc >> 7, d0 = (oc & 127) + half8;
      const float* eh = emb_h + (n >> 5) * DQK + d0;
      const float* ew = emb_w + (n & 31) * DQK + d0;
      v4u pk;
#pragma unroll
      for (int j = 0; j < 4; ++j) {
        float e0 = acc[t][2 * j]     + eh[2 * j]     + ew[2 * j];
        float e1 = acc[t][2 * j + 1] + eh[2 * j + 1] + ew[2 * j + 1];
        pk[j] = (unsigned)f2bf(e0) | ((unsigned)f2bf(e1) << 16);
      }
      *(v4u*)(Kp + ((size_t)(b * HEADS + head) * NN + n) * DQK + d0) = pk;
    } else {                                    // V: [bh][d][n]
      int oc = (ot - 64) * 16, head = oc >> 7, d0 = (oc & 127) + half8;
#pragma unroll
      for (int j = 0; j < 8; ++j)
        Vt[((size_t)(b * HEADS + head) * DV + d0 + j) * NN + n] = f2bf(acc[t][j]);
    }
  }
}

// ---------------- flash attention with TDM-staged K/V tiles ----------------
__global__ void k_attn(const unsigned short* __restrict__ Qs,
                       const unsigned short* __restrict__ Kp,
                       const unsigned short* __restrict__ Vt,
                       float* __restrict__ out) {
  __shared__ __align__(16) unsigned short KT[64 * KT_PITCH];       // K' chunk [y][d]
  __shared__ __align__(16) unsigned short VT[128 * VT_PITCH];      // V  chunk [d][y]
  __shared__ __align__(16) unsigned short PL[8 * 16 * PL_PITCH];   // per-wave P tiles

  int lane = threadIdx.x & 31;
  int w    = threadIdx.x >> 5;
  int bh   = blockIdx.x >> 3;                       // 0..63
  int x0   = ((blockIdx.x & 7) * 8 + w) * 16;       // query row block
  int col  = lane & 15;
  int half = lane >> 4;

  const unsigned short* qrow = Qs + ((size_t)bh * NN + x0 + col) * DQK;
  v16bf qf[4];
#pragma unroll
  for (int k = 0; k < 4; ++k) qf[k] = load_a(qrow, k * 32, lane);

  v8f acc[8] = {};
  float m[8], lsum[8];
#pragma unroll
  for (int j = 0; j < 8; ++j) { m[j] = -3.0e38f; lsum[j] = 0.f; }

  const unsigned short* kbase = Kp + (size_t)bh * NN * DQK;
  const unsigned short* vbase = Vt + (size_t)bh * DV * NN;
  unsigned short* mypl = PL + w * (16 * PL_PITCH);

  for (int yc = 0; yc < 16; ++yc) {
    int y0 = yc * 64;
    __syncthreads();                               // previous chunk fully consumed
#if HAVE_TDM
    if (w == 0) {
      // K': tile 128(d) x 64(y rows), pitch 128; pad 4 dwords every 64 dwords
      tdm_load_2d((unsigned)(uintptr_t)KT, kbase + (size_t)y0 * DQK,
                  128u, 64u, 128ull, 128u, 64u, 5u, 3u);
      // V: tile 64(y) x 128(d rows), pitch NN; pad 4 dwords every 32 dwords
      tdm_load_2d((unsigned)(uintptr_t)VT, vbase + y0,
                  64u, 128u, (unsigned long long)NN, 1024u, 128u, 4u, 3u);
      __builtin_amdgcn_s_wait_tensorcnt(0);
    }
#else
    for (int i = threadIdx.x; i < 64 * 16; i += 256) {      // K chunk, 16B units
      int r = i >> 4, c8 = (i & 15) << 3;
      *(v4u*)(KT + r * KT_PITCH + c8) = *(const v4u*)(kbase + (size_t)(y0 + r) * DQK + c8);
    }
    for (int i = threadIdx.x; i < 128 * 8; i += 256) {      // V chunk, 16B units
      int r = i >> 3, c8 = (i & 7) << 3;
      *(v4u*)(VT + r * VT_PITCH + c8) = *(const v4u*)(vbase + (size_t)r * NN + y0 + c8);
    }
#endif
    __syncthreads();

    // S = Q K'^T  (K' fragments from LDS)
    v8f S[4] = {};
#pragma unroll
    for (int t = 0; t < 4; ++t) {
      const unsigned short* krow = KT + (t * 16 + col) * KT_PITCH;
#pragma unroll
      for (int k = 0; k < 4; ++k)
        S[t] = wmma_bf16(qf[k], load_b(krow, k * 32, lane), S[t]);
    }
    // online softmax update (rows at VGPR j, half-group `half`)
#pragma unroll
    for (int j = 0; j < 8; ++j) {
      float v = fmaxf(fmaxf(S[0][j], S[1][j]), fmaxf(S[2][j], S[3][j]));
      v = fmaxf(v, __shfl_xor(v, 1, 16));
      v = fmaxf(v, __shfl_xor(v, 2, 16));
      v = fmaxf(v, __shfl_xor(v, 4, 16));
      v = fmaxf(v, __shfl_xor(v, 8, 16));
      float mn    = fmaxf(m[j], v);
      float alpha = __expf(m[j] - mn);
      m[j] = mn;
      lsum[j] *= alpha;
#pragma unroll
      for (int dt = 0; dt < 8; ++dt) acc[dt][j] *= alpha;
      float rs = 0.f;
#pragma unroll
      for (int t = 0; t < 4; ++t) {
        float p = __expf(S[t][j] - mn);
        rs += p;
        mypl[(j + 8 * half) * PL_PITCH + t * 16 + col] = f2bf(p);
      }
      rs += __shfl_xor(rs, 1, 16);
      rs += __shfl_xor(rs, 2, 16);
      rs += __shfl_xor(rs, 4, 16);
      rs += __shfl_xor(rs, 8, 16);
      lsum[j] += rs;
    }
    // out += P @ V  (both operands from LDS)
#pragma unroll
    for (int kt = 0; kt < 2; ++kt) {
      v16bf pf = load_a(mypl + col * PL_PITCH, kt * 32, lane);
#pragma unroll
      for (int dt = 0; dt < 8; ++dt) {
        const unsigned short* vrow = VT + (dt * 16 + col) * VT_PITCH;
        acc[dt] = wmma_bf16(pf, load_b(vrow, kt * 32, lane), acc[dt]);
      }
    }
  }
  // epilogue: out[bh][d][x] = acc / lsum  (transposed D store, coalesced)
  float* obase = out + (size_t)bh * DV * NN;
#pragma unroll
  for (int dt = 0; dt < 8; ++dt) {
    float* op = obase + (size_t)(dt * 16 + col) * NN + x0 + 8 * half;
    v4f o0, o1;
#pragma unroll
    for (int j = 0; j < 4; ++j) o0[j] = acc[dt][j] / lsum[j];
#pragma unroll
    for (int j = 0; j < 4; ++j) o1[j] = acc[dt][4 + j] / lsum[4 + j];
    *(v4f*)op       = o0;
    *(v4f*)(op + 4) = o1;
  }
}

extern "C" void kernel_launch(void* const* d_in, const int* in_sizes, int n_in,
                              void* d_out, int out_size, void* d_ws, size_t ws_size,
                              hipStream_t stream) {
  const float* fm   = (const float*)d_in[0];
  const float* wqk  = (const float*)d_in[1];
  const float* wv   = (const float*)d_in[2];
  const float* embh = (const float*)d_in[3];
  const float* embw = (const float*)d_in[4];
  float* out = (float*)d_out;

  char* ws = (char*)d_ws;
  unsigned short* Xt  = (unsigned short*)(ws);             // 16 MB  [B][N][C] bf16
  unsigned short* Wqk = (unsigned short*)(ws + 16777216);  // 1 MB
  unsigned short* Wv  = (unsigned short*)(ws + 17825792);  // 0.5 MB
  unsigned short* Qs  = (unsigned short*)(ws + 18350080);  // 16 MB  [bh][n][d]
  unsigned short* Kp  = (unsigned short*)(ws + 35127296);  // 16 MB  [bh][n][d]
  unsigned short* Vt  = (unsigned short*)(ws + 51904512);  // 16 MB  [bh][d][n]

  k_transpose<<<dim3(32, 16, 16), dim3(32, 8), 0, stream>>>(fm, Xt);
  k_wconv<<<3072, 256, 0, stream>>>(wqk, wv, Wqk, Wv);
  k_proj<<<3072, 256, 0, stream>>>(Xt, Wqk, Wv, embh, embw, Qs, Kp, Vt);
  k_attn<<<512, 256, 0, stream>>>(Qs, Kp, Vt, out);
}
